// RGCN_83476984365514
// MI455X (gfx1250) — matrix operand order repaired and checked by reading, not compiled, not run on hardware
//
#include <hip/hip_runtime.h>
#include <hip/hip_bf16.h>

#define INF   64
#define HIDF  128
#define OUTF  32
#define HEADS 16
#define DH    8
#define NEG_SLOPE 0.2f
#define BN_EPS 1e-5f

typedef __bf16 bf16;
typedef __attribute__((ext_vector_type(16))) __bf16 v16bf;
typedef __attribute__((ext_vector_type(8)))  __bf16 v8bf;
typedef __attribute__((ext_vector_type(8)))  float  v8f;

// ---------------------------------------------------------------------------
// Weight convert + transpose: Wt[m*K + k] = (bf16) W[k*M + m]
// ---------------------------------------------------------------------------
__global__ void wconv_kernel(const float* __restrict__ W, bf16* __restrict__ Wt,
                             int K, int M) {
    int t = blockIdx.x * blockDim.x + threadIdx.x;
    if (t >= K * M) return;
    int m = t / K, k = t % K;
    Wt[t] = (bf16)W[(size_t)k * M + m];
}

// ---------------------------------------------------------------------------
// WMMA GEMM: C[N x Mout] = A[N x Ktot] * Wt^T   (Wt stored [Mout x Ktot] bf16)
// Mout = NT*16, Ktot = KT*32 (compile time). Optional residual add.
// Block = 128 threads = 4 waves; each wave computes a 16 x Mout strip.
// Wt is staged into LDS via async global->LDS copies (ASYNCcnt), B fragments
// then come from LDS (ds_load_b128, 16-lane conflict-free via +8 elem pad).
// A loads are software-pipelined one k-tile ahead of the WMMAs.
// ---------------------------------------------------------------------------
template <int NT, int KT>
__global__ __launch_bounds__(128)
void gemm_kernel(const float* __restrict__ A, const bf16* __restrict__ Wt,
                 const float* __restrict__ resid, float* __restrict__ C,
                 int Nrows) {
    constexpr int Mout   = NT * 16;
    constexpr int Ktot   = KT * 32;
    constexpr int STRIDE = Ktot + 8;          // +16B pad: bank shift 4 per row

    __shared__ bf16 sW[Mout * STRIDE];

    // ---- async stage Wt[Mout x Ktot] into LDS (16B chunks, all lanes active)
    {
        constexpr int CHUNKS = Mout * (Ktot / 8);   // multiples of 128
        for (int idx = threadIdx.x; idx < CHUNKS; idx += 128) {
            int row = idx / (Ktot / 8);
            int ch  = idx % (Ktot / 8);
            unsigned ldsaddr =
                (unsigned)(uintptr_t)(&sW[row * STRIDE + ch * 8]);
            const bf16* gp = Wt + (size_t)row * Ktot + ch * 8;
            asm volatile("global_load_async_to_lds_b128 %0, %1, off"
                         :: "v"(ldsaddr), "v"(gp) : "memory");
        }
        asm volatile("s_wait_asynccnt 0x0" ::: "memory");
    }
    __syncthreads();

    const int lane = threadIdx.x & 31;
    const int wave = threadIdx.x >> 5;
    const int half = lane >> 4;   // A frag: 0 -> K 0-7 & 16-23, 1 -> 8-15 & 24-31
    const int r    = lane & 15;
    const int rowBase = blockIdx.x * 64 + wave * 16;

    v8f acc[NT] = {};

    int arow = rowBase + r;                 // A-fragment row (M = lane%16)
    if (arow > Nrows - 1) arow = Nrows - 1; // clamp: EXEC must stay all-ones
    const float* Ap = A + (size_t)arow * Ktot;

    v8f x0 = *(const v8f*)(Ap + half * 8);
    v8f x1 = *(const v8f*)(Ap + 16 + half * 8);

    #pragma unroll
    for (int kt = 0; kt < KT; kt++) {
        v16bf a;
        #pragma unroll
        for (int e = 0; e < 8; e++) {
            a[e]     = (bf16)x0[e];
            a[8 + e] = (bf16)x1[e];
        }
        if (kt + 1 < KT) {      // prefetch next k-tile while WMMAs run
            x0 = *(const v8f*)(Ap + (kt + 1) * 32 + half * 8);
            x1 = *(const v8f*)(Ap + (kt + 1) * 32 + 16 + half * 8);
        }
        #pragma unroll
        for (int t = 0; t < NT; t++) {
            const bf16* Bp = &sW[(t * 16 + r) * STRIDE + kt * 32 + half * 8];
            v8bf blo = *(const v8bf*)Bp;        // 16B-aligned ds_load_b128
            v8bf bhi = *(const v8bf*)(Bp + 16);
            v16bf b;
            #pragma unroll
            for (int e = 0; e < 8; e++) { b[e] = blo[e]; b[8 + e] = bhi[e]; }
            acc[t] = __builtin_amdgcn_wmma_f32_16x16x32_bf16(
                false, a, false, b, (short)0, acc[t], false, false);
        }
    }

    #pragma unroll
    for (int t = 0; t < NT; t++) {
        int col = t * 16 + r;
        #pragma unroll
        for (int i = 0; i < 8; i++) {
            int row = rowBase + half * 8 + i;   // C frag: M = i + 8*half
            if (row < Nrows) {
                float v = acc[t][i];
                if (resid) v += resid[(size_t)row * Mout + col];
                C[(size_t)row * Mout + col] = v;
            }
        }
    }
}

// ---------------------------------------------------------------------------
// BatchNorm statistics: stats[0..127] += sum, stats[128..255] += sum of sq
// ---------------------------------------------------------------------------
__global__ void bn_stats_kernel(const float* __restrict__ x,
                                float* __restrict__ stats, int nrows) {
    int c = threadIdx.x & (HIDF - 1);
    int rstart  = blockIdx.x * (blockDim.x / HIDF) + (threadIdx.x >> 7);
    int rstride = gridDim.x * (blockDim.x / HIDF);
    float s = 0.f, q = 0.f;
    for (int row = rstart; row < nrows; row += rstride) {
        float v = x[(size_t)row * HIDF + c];
        s += v; q += v * v;
    }
    atomicAdd(&stats[c], s);
    atomicAdd(&stats[HIDF + c], q);
}

__global__ void bn_relu_kernel(const float* __restrict__ x,
                               const float* __restrict__ stats,
                               const float* __restrict__ g,
                               const float* __restrict__ b,
                               float* __restrict__ y, int nrows) {
    int t = blockIdx.x * blockDim.x + threadIdx.x;
    if (t >= nrows * HIDF) return;
    int c = t & (HIDF - 1);
    float inv_n = 1.f / (float)nrows;
    float mean = stats[c] * inv_n;
    float var  = stats[HIDF + c] * inv_n - mean * mean;  // biased var
    float v = g[c] * (x[t] - mean) * rsqrtf(var + BN_EPS) + b[c];
    y[t] = fmaxf(v, 0.f);
}

// ---------------------------------------------------------------------------
// GAT attention scores: el/er[n,h] = sum_d f[n,h,d] * al/ar[h,d]
// ---------------------------------------------------------------------------
__global__ void attn_scores_kernel(const float* __restrict__ f,
                                   const float* __restrict__ al,
                                   const float* __restrict__ ar,
                                   float* __restrict__ el,
                                   float* __restrict__ er, int n) {
    int t = blockIdx.x * blockDim.x + threadIdx.x;
    if (t >= n * HEADS) return;
    int node = t >> 4, h = t & 15;
    const float* fp = f + (size_t)node * HIDF + h * DH;
    float sl = 0.f, sr = 0.f;
    #pragma unroll
    for (int d = 0; d < DH; d++) {
        float v = fp[d];
        sl += v * al[h * DH + d];
        sr += v * ar[h * DH + d];
    }
    el[t] = sl; er[t] = sr;
}

// monotone float <-> uint mapping for atomic max (init encoded value = 0)
__device__ __forceinline__ unsigned int fenc(float f) {
    unsigned int u = __float_as_uint(f);
    return (u & 0x80000000u) ? ~u : (u | 0x80000000u);
}
__device__ __forceinline__ float fdec(unsigned int u) {
    return __uint_as_float((u & 0x80000000u) ? (u & 0x7FFFFFFFu) : ~u);
}

__global__ void edge_max_kernel(const int* __restrict__ src,
                                const int* __restrict__ dst,
                                const float* __restrict__ el,
                                const float* __restrict__ er,
                                unsigned int* __restrict__ menc, int E) {
    int t = blockIdx.x * blockDim.x + threadIdx.x;
    if (t >= E * HEADS) return;
    int e = t >> 4, h = t & 15;
    float v = el[(size_t)src[e] * HEADS + h] + er[(size_t)dst[e] * HEADS + h];
    v = v > 0.f ? v : NEG_SLOPE * v;
    atomicMax(&menc[(size_t)dst[e] * HEADS + h], fenc(v));
}

__global__ void edge_exp_kernel(const int* __restrict__ src,
                                const int* __restrict__ dst,
                                const float* __restrict__ el,
                                const float* __restrict__ er,
                                const unsigned int* __restrict__ menc,
                                float* __restrict__ exbuf,
                                float* __restrict__ ssum, int E) {
    int t = blockIdx.x * blockDim.x + threadIdx.x;
    if (t >= E * HEADS) return;
    int e = t >> 4, h = t & 15;
    int d = dst[e];
    float v = el[(size_t)src[e] * HEADS + h] + er[(size_t)d * HEADS + h];
    v = v > 0.f ? v : NEG_SLOPE * v;
    float m  = fdec(menc[(size_t)d * HEADS + h]);
    float ex = __expf(v - m);
    exbuf[t] = ex;
    atomicAdd(&ssum[(size_t)d * HEADS + h], ex);
}

__global__ void edge_aggr_kernel(const int* __restrict__ src,
                                 const int* __restrict__ dst,
                                 const float* __restrict__ exbuf,
                                 const float* __restrict__ ssum,
                                 const float* __restrict__ f,
                                 float* __restrict__ g, int E) {
    int t = blockIdx.x * blockDim.x + threadIdx.x;
    if (t >= E * HEADS) return;
    int e = t >> 4, h = t & 15;
    int d = dst[e], s = src[e];
    float a = exbuf[t] / ssum[(size_t)d * HEADS + h];
    const float* fp = f + (size_t)s * HIDF + h * DH;
    float*       gp = g + (size_t)d * HIDF + h * DH;
    #pragma unroll
    for (int j = 0; j < DH; j++) atomicAdd(&gp[j], a * fp[j]);
}

// h = relu(g + bias) + h   (residual, in place)
__global__ void gat_finalize_kernel(const float* __restrict__ g,
                                    const float* __restrict__ bias,
                                    float* __restrict__ h, int nrows) {
    int t = blockIdx.x * blockDim.x + threadIdx.x;
    if (t >= nrows * HIDF) return;
    int c = t & (HIDF - 1);
    h[t] = fmaxf(g[t] + bias[c], 0.f) + h[t];
}

// ---------------------------------------------------------------------------
extern "C" void kernel_launch(void* const* d_in, const int* in_sizes, int n_in,
                              void* d_out, int out_size, void* d_ws, size_t ws_size,
                              hipStream_t stream) {
    const int N = in_sizes[0] / INF;   // 50000
    const int E = in_sizes[2];         // 500000
    (void)n_in; (void)out_size; (void)ws_size;

    const float* node1 = (const float*)d_in[0];
    const float* node2 = (const float*)d_in[1];
    const int* srcs[2] = {(const int*)d_in[2], (const int*)d_in[4]};
    const int* dsts[2] = {(const int*)d_in[3], (const int*)d_in[5]};
    // params (depth-first dict order):
    //   6+r*4:  emb{r}.{W1,g,b,W2}
    //   14+(l*2+r)*4: gat{l}_{r}.{W,al,ar,bias}
    //   46..49: dec.{W1,g,b,W2}

    char* ws = (char*)d_ws;
    size_t off = 0;
    auto alloc = [&](size_t bytes) -> void* {
        void* p = ws + off;
        off += (bytes + 255) & ~(size_t)255;
        return p;
    };

    bf16 *embW1t[2], *embW2t[2], *gatWt[8];
    for (int r = 0; r < 2; r++) {
        embW1t[r] = (bf16*)alloc((size_t)INF * HIDF * 2);
        embW2t[r] = (bf16*)alloc((size_t)HIDF * HIDF * 2);
    }
    for (int i = 0; i < 8; i++) gatWt[i] = (bf16*)alloc((size_t)HIDF * HIDF * 2);
    bf16* decW1ta = (bf16*)alloc((size_t)HIDF * HIDF * 2);  // rows k 0..127
    bf16* decW1tb = (bf16*)alloc((size_t)HIDF * HIDF * 2);  // rows k 128..255
    bf16* decW2t  = (bf16*)alloc((size_t)HIDF * OUTF * 2);

    float* h[2];
    h[0] = (float*)alloc((size_t)N * HIDF * 4);
    h[1] = (float*)alloc((size_t)N * HIDF * 4);
    float* tbuf = (float*)alloc((size_t)N * HIDF * 4);
    float* gbuf = (float*)alloc((size_t)N * HIDF * 4);
    float* el   = (float*)alloc((size_t)N * HEADS * 4);
    float* er   = (float*)alloc((size_t)N * HEADS * 4);
    unsigned int* menc = (unsigned int*)alloc((size_t)N * HEADS * 4);
    float* ssum = (float*)alloc((size_t)N * HEADS * 4);
    float* exbuf = (float*)alloc((size_t)E * HEADS * 4);
    float* stats = (float*)alloc(256 * 4);

    auto wconv = [&](const float* W, bf16* Wt, int K, int M) {
        int n = K * M;
        wconv_kernel<<<(n + 255) / 256, 256, 0, stream>>>(W, Wt, K, M);
    };
    for (int r = 0; r < 2; r++) {
        wconv((const float*)d_in[6 + r * 4 + 0], embW1t[r], INF,  HIDF);
        wconv((const float*)d_in[6 + r * 4 + 3], embW2t[r], HIDF, HIDF);
    }
    for (int i = 0; i < 8; i++)
        wconv((const float*)d_in[14 + i * 4 + 0], gatWt[i], HIDF, HIDF);
    const float* decW1 = (const float*)d_in[46];           // [256 x 128]
    wconv(decW1,               decW1ta, HIDF, HIDF);
    wconv(decW1 + HIDF * HIDF, decW1tb, HIDF, HIDF);
    wconv((const float*)d_in[49], decW2t, HIDF, OUTF);

    const int gemmBlocks = (N + 63) / 64;
    const int nh = N * HEADS, eh = E * HEADS;
    const int nf = N * HIDF;

    // ---- per-relation embed MLP2: h_r = relu(BN(x W1)) W2 + relu(BN(x W1))
    for (int r = 0; r < 2; r++) {
        const float* x = (r == 0) ? node1 : node2;
        gemm_kernel<8, 2><<<gemmBlocks, 128, 0, stream>>>(x, embW1t[r],
                                                          nullptr, tbuf, N);
        hipMemsetAsync(stats, 0, 256 * 4, stream);
        bn_stats_kernel<<<256, 256, 0, stream>>>(tbuf, stats, N);
        bn_relu_kernel<<<(nf + 255) / 256, 256, 0, stream>>>(
            tbuf, stats, (const float*)d_in[6 + r * 4 + 1],
            (const float*)d_in[6 + r * 4 + 2], gbuf, N);
        gemm_kernel<8, 4><<<gemmBlocks, 128, 0, stream>>>(gbuf, embW2t[r],
                                                          gbuf, h[r], N);
    }

    // ---- GAT layers
    for (int l = 0; l < 4; l++) {
        for (int r = 0; r < 2; r++) {
            int gi = l * 2 + r;
            const float* al   = (const float*)d_in[14 + gi * 4 + 1];
            const float* ar   = (const float*)d_in[14 + gi * 4 + 2];
            const float* bias = (const float*)d_in[14 + gi * 4 + 3];
            gemm_kernel<8, 4><<<gemmBlocks, 128, 0, stream>>>(h[r], gatWt[gi],
                                                              nullptr, tbuf, N);
            attn_scores_kernel<<<(nh + 255) / 256, 256, 0, stream>>>(tbuf, al, ar,
                                                                     el, er, N);
            hipMemsetAsync(menc, 0, (size_t)N * HEADS * 4, stream);
            hipMemsetAsync(ssum, 0, (size_t)N * HEADS * 4, stream);
            hipMemsetAsync(gbuf, 0, (size_t)N * HIDF * 4, stream);
            edge_max_kernel<<<(eh + 255) / 256, 256, 0, stream>>>(
                srcs[r], dsts[r], el, er, menc, E);
            edge_exp_kernel<<<(eh + 255) / 256, 256, 0, stream>>>(
                srcs[r], dsts[r], el, er, menc, exbuf, ssum, E);
            edge_aggr_kernel<<<(eh + 255) / 256, 256, 0, stream>>>(
                srcs[r], dsts[r], exbuf, ssum, tbuf, gbuf, E);
            gat_finalize_kernel<<<(nf + 255) / 256, 256, 0, stream>>>(gbuf, bias,
                                                                      h[r], N);
        }
    }

    // ---- decision MLP on concat(h0, h1): split-K into two chained GEMMs
    gemm_kernel<8, 4><<<gemmBlocks, 128, 0, stream>>>(h[0], decW1ta,
                                                      nullptr, tbuf, N);
    gemm_kernel<8, 4><<<gemmBlocks, 128, 0, stream>>>(h[1], decW1tb,
                                                      tbuf, tbuf, N);
    hipMemsetAsync(stats, 0, 256 * 4, stream);
    bn_stats_kernel<<<256, 256, 0, stream>>>(tbuf, stats, N);
    bn_relu_kernel<<<(nf + 255) / 256, 256, 0, stream>>>(
        tbuf, stats, (const float*)d_in[47], (const float*)d_in[48], gbuf, N);
    gemm_kernel<2, 4><<<gemmBlocks, 128, 0, stream>>>(gbuf, decW2t,
                                                      nullptr, (float*)d_out, N);
}
